// AttentionHeads_52269751992484
// MI455X (gfx1250) — matrix operand order repaired and checked by reading, not compiled, not run on hardware
//
#include <hip/hip_runtime.h>
#include <hip/hip_bf16.h>

typedef __attribute__((ext_vector_type(16))) __bf16 v16bf;
typedef __attribute__((ext_vector_type(8)))  __bf16 v8bf;
typedef __attribute__((ext_vector_type(4)))  __bf16 v4bf;
typedef __attribute__((ext_vector_type(2)))  __bf16 v2bf;
typedef __attribute__((ext_vector_type(8)))  float  v8f;

#define B_  8
#define T_  1024
#define C_  1024
#define NH_ 16
#define DH_ 64
#define M_  (B_ * T_)   // 8192

static __device__ __forceinline__ v8f wmma_bf16(v16bf a, v16bf b, v8f c) {
    return __builtin_amdgcn_wmma_f32_16x16x32_bf16(false, a, false, b, (short)0, c,
                                                   false, false);
}

// ---------------------------------------------------------------------------
// Prep 0: x (f32 [M][C]) -> xb (bf16 [M][C]).  grid = M, block = 256.
// ---------------------------------------------------------------------------
__global__ __launch_bounds__(256)
void conv_x_kernel(const float* __restrict__ x, __bf16* __restrict__ xb)
{
    size_t e = ((size_t)blockIdx.x * 256 + threadIdx.x) * 4;
    float4 f = *(const float4*)(x + e);
    v4bf o = { (__bf16)f.x, (__bf16)f.y, (__bf16)f.z, (__bf16)f.w };
    *(v4bf*)(xb + e) = o;
}

// ---------------------------------------------------------------------------
// Prep 1: W (f32 [K][N]) -> Wt (bf16 [N][K]) for Wq/Wk/Wv. grid=(4096,1,3).
// ---------------------------------------------------------------------------
__global__ __launch_bounds__(256)
void trans_w_kernel(const float* __restrict__ Wq, const float* __restrict__ Wk,
                    const float* __restrict__ Wv, __bf16* __restrict__ Wt)
{
    const float* W = (blockIdx.z == 0) ? Wq : (blockIdx.z == 1) ? Wk : Wv;
    __bf16* dst = Wt + (size_t)blockIdx.z * C_ * C_;
    size_t e = (size_t)blockIdx.x * 256 + threadIdx.x;   // 0 .. 1M-1
    int k = (int)(e >> 10);
    int n = (int)(e & (C_ - 1));
    dst[(size_t)n * C_ + k] = (__bf16)W[e];
}

// ---------------------------------------------------------------------------
// Kernel 1: fused QKV projection, LDS-free. q = x @ W + b.
// A-fragments from bf16 x rows (contiguous), B-fragments from W^T rows
// (contiguous). grid = (M/128, NH, 3), block = 256 (8 waves, 16x64 each).
// Q,K stored bf16 [B][NH][T][DH]; V stored transposed [B][NH][DH][T].
// ---------------------------------------------------------------------------
__global__ __launch_bounds__(256)
void qkv_gemm_kernel(const __bf16* __restrict__ xb, const __bf16* __restrict__ Wt3,
                     const float* __restrict__ bq, const float* __restrict__ bk,
                     const float* __restrict__ bv,
                     __bf16* __restrict__ Qd, __bf16* __restrict__ Kd,
                     __bf16* __restrict__ Vtd)
{
    const int tid  = threadIdx.x;
    const int wave = tid >> 5;
    const int lane = tid & 31;
    const int hi   = lane >> 4;
    const int lo   = lane & 15;

    const int m0 = blockIdx.x * 128;
    const int h  = blockIdx.y;
    const int n0 = h * DH_;
    const int z  = blockIdx.z;

    const __bf16* Wt  = Wt3 + (size_t)z * C_ * C_;
    const float*  bias = (z == 0) ? bq : (z == 1) ? bk : bv;

    // per-lane operand row pointers
    const __bf16* arow = xb + (size_t)(m0 + wave * 16 + lo) * C_;
    const __bf16* brow[4];
    #pragma unroll
    for (int nt = 0; nt < 4; ++nt)
        brow[nt] = Wt + (size_t)(n0 + nt * 16 + lo) * C_ + 16 * hi;

    v8f acc[4] = {};

    for (int k0 = 0; k0 < C_; k0 += 32) {
        __builtin_prefetch((const void*)(arow + k0 + 64), 0, 1);
        // A fragment: two contiguous 16B runs from this lane's x row
        v8bf a0 = *(const v8bf*)(arow + k0 + 16 * hi);
        v8bf a1 = *(const v8bf*)(arow + k0 + 16 * hi + 8);
        v16bf afrag;
        #pragma unroll
        for (int j = 0; j < 8; ++j) { afrag[j] = a0[j]; afrag[8 + j] = a1[j]; }
        // B fragments: contiguous 32B from W^T rows
        #pragma unroll
        for (int nt = 0; nt < 4; ++nt) {
            v16bf bfrag = *(const v16bf*)(brow[nt] + k0);
            acc[nt] = wmma_bf16(afrag, bfrag, acc[nt]);
        }
    }

    // ---- epilogue ----
    const int trow = m0 + wave * 16 + 8 * hi;   // row base for elements r=0..7
    const int bb = trow >> 10;                  // batch
    const int tt = trow & (T_ - 1);             // t base (multiple of 8)

    if (z == 2) {
        // V: transposed store [B][NH][DH][T]; element r -> contiguous t
        #pragma unroll
        for (int nt = 0; nt < 4; ++nt) {
            float bvv = bias[n0 + nt * 16 + lo];
            int d = n0 + nt * 16 + lo;
            v8bf pk;
            #pragma unroll
            for (int r = 0; r < 8; ++r) pk[r] = (__bf16)(acc[nt][r] + bvv);
            size_t idx = ((size_t)bb * NH_ * DH_ + d) * T_ + tt;  // d = h*64+..
            *(v8bf*)(Vtd + idx) = pk;
        }
    } else {
        __bf16* dst = (z == 0) ? Qd : Kd;
        #pragma unroll
        for (int nt = 0; nt < 4; ++nt) {
            float bvv = bias[n0 + nt * 16 + lo];
            #pragma unroll
            for (int r = 0; r < 8; ++r) {
                size_t idx = ((size_t)(bb * NH_ + h) * T_ + tt + r) * DH_
                           + nt * 16 + lo;
                dst[idx] = (__bf16)(acc[nt][r] + bvv);
            }
        }
    }
}

// ---------------------------------------------------------------------------
// Kernel 2: flash-style non-causal attention per (b,h), barrier-free.
// grid = (T/128, NH, B), block = 256 (8 waves); each wave owns 16 query rows.
// K fragments: direct 32B loads from K rows. V fragments: direct 32B loads
// from V^T rows. P re-layout via per-wave LDS (packed even/odd key columns).
// ---------------------------------------------------------------------------
__global__ __launch_bounds__(256)
void attn_kernel(const __bf16* __restrict__ Qw, const __bf16* __restrict__ Kw,
                 const __bf16* __restrict__ Vtw, float* __restrict__ out)
{
    __shared__ __align__(32) __bf16 Pl[8][16][32];   // per-wave P scratch, 8 KB

    const int tid  = threadIdx.x;
    const int wave = tid >> 5;
    const int lane = tid & 31;
    const int hi   = lane >> 4;
    const int lo   = lane & 15;

    const int h  = blockIdx.y;
    const int b  = blockIdx.z;
    const int q0 = blockIdx.x * 128 + wave * 16;

    const size_t base = (size_t)(b * NH_ + h) * T_ * DH_;
    const __bf16* Q  = Qw + base;
    const __bf16* K  = Kw + base;
    const __bf16* Vt = Vtw + base;   // [DH][T] for this (b,h)

    // Q fragments (two 16x32 k-steps covering DH=64): 2x16B vector loads each
    v16bf qa[2];
    #pragma unroll
    for (int s = 0; s < 2; ++s) {
        const __bf16* qrow = Q + (size_t)(q0 + lo) * DH_ + s * 32 + 16 * hi;
        v8bf qlo8 = *(const v8bf*)(qrow);
        v8bf qhi8 = *(const v8bf*)(qrow + 8);
        #pragma unroll
        for (int j = 0; j < 8; ++j) { qa[s][j] = qlo8[j]; qa[s][8 + j] = qhi8[j]; }
    }

    // per-lane V^T row pointers (fixed dh column per lane, contiguous keys)
    const __bf16* vrow[4];
    #pragma unroll
    for (int nt = 0; nt < 4; ++nt)
        vrow[nt] = Vt + (size_t)(nt * 16 + lo) * T_ + 16 * hi;

    v8f o[4] = {};
    float mrun[8], lrun[8];
    #pragma unroll
    for (int r = 0; r < 8; ++r) { mrun[r] = -1e30f; lrun[r] = 0.0f; }

    const float scale = 0.125f;   // 1/sqrt(64)

    for (int kb = 0; kb < T_; kb += 32) {
        __builtin_prefetch((const void*)(vrow[0] + kb + 32), 0, 1);
        // ---- S tiles: tile0 = even keys kb+2*lo, tile1 = odd keys kb+2*lo+1
        const v16bf* krow0 = (const v16bf*)(K + (size_t)(kb + 2 * lo) * DH_);
        const v16bf* krow1 = (const v16bf*)(K + (size_t)(kb + 2 * lo + 1) * DH_);
        v16bf kb00 = krow0[hi];       // kd  0..31
        v16bf kb01 = krow0[2 + hi];   // kd 32..63
        v16bf kb10 = krow1[hi];
        v16bf kb11 = krow1[2 + hi];

        v8f s0 = {}, s1 = {};
        s0 = wmma_bf16(qa[0], kb00, s0);
        s0 = wmma_bf16(qa[1], kb01, s0);
        s1 = wmma_bf16(qa[0], kb10, s1);
        s1 = wmma_bf16(qa[1], kb11, s1);

        // ---- online softmax (f32); row r+8*hi lives in element r across lanes
        #pragma unroll
        for (int r = 0; r < 8; ++r) {
            float a0 = s0[r] * scale;     // key kb + 2*lo
            float a1 = s1[r] * scale;     // key kb + 2*lo + 1
            float mx = fmaxf(a0, a1);
            #pragma unroll
            for (int off = 1; off < 16; off <<= 1)
                mx = fmaxf(mx, __shfl_xor(mx, off, 32));
            float mn   = fmaxf(mrun[r], mx);
            float corr = __expf(mrun[r] - mn);
            float p0   = __expf(a0 - mn);
            float p1   = __expf(a1 - mn);
            float rs   = p0 + p1;
            #pragma unroll
            for (int off = 1; off < 16; off <<= 1)
                rs += __shfl_xor(rs, off, 32);
            lrun[r] = lrun[r] * corr + rs;
            mrun[r] = mn;
            #pragma unroll
            for (int nt = 0; nt < 4; ++nt) o[nt][r] *= corr;
            // adjacent even/odd keys -> one packed 32-bit LDS store
            v2bf pp = { (__bf16)p0, (__bf16)p1 };
            *(v2bf*)(&Pl[wave][r + 8 * hi][2 * lo]) = pp;
        }

        // ---- P (16x32, key-local = identity) -> A-fragment layout ----
        v16bf pa;
        #pragma unroll
        for (int j = 0; j < 16; ++j) {
            int kl = j + 16 * hi;
            pa[j] = Pl[wave][lo][kl];
        }

        // ---- O += P V : 4 dh tiles, V^T rows are contiguous 32B loads ----
        #pragma unroll
        for (int nt = 0; nt < 4; ++nt) {
            v16bf vb = *(const v16bf*)(vrow[nt] + kb);
            o[nt] = wmma_bf16(pa, vb, o[nt]);
        }
    }

    // ---- normalize and store f32 output [B][T][C] ----
    #pragma unroll
    for (int nt = 0; nt < 4; ++nt) {
        #pragma unroll
        for (int r = 0; r < 8; ++r) {
            int t = q0 + r + 8 * hi;
            float val = o[nt][r] / lrun[r];
            out[((size_t)b * T_ + t) * C_ + h * DH_ + nt * 16 + lo] = val;
        }
    }
}

extern "C" void kernel_launch(void* const* d_in, const int* in_sizes, int n_in,
                              void* d_out, int out_size, void* d_ws, size_t ws_size,
                              hipStream_t stream) {
    const float* x  = (const float*)d_in[0];
    const float* Wq = (const float*)d_in[1];
    const float* bq = (const float*)d_in[2];
    const float* Wk = (const float*)d_in[3];
    const float* bk = (const float*)d_in[4];
    const float* Wv = (const float*)d_in[5];
    const float* bv = (const float*)d_in[6];
    float* out = (float*)d_out;

    // workspace layout (bf16): Q | K | Vt | xb | Wt(x3)  = 70 MB
    const size_t per = (size_t)B_ * NH_ * T_ * DH_;   // 8.39M elems
    __bf16* Qd  = (__bf16*)d_ws;
    __bf16* Kd  = Qd + per;
    __bf16* Vtd = Kd + per;
    __bf16* xbd = Vtd + per;
    __bf16* Wtd = xbd + (size_t)M_ * C_;

    conv_x_kernel<<<dim3(M_ * C_ / 1024), dim3(256), 0, stream>>>(x, xbd);
    trans_w_kernel<<<dim3(C_ * C_ / 256, 1, 3), dim3(256), 0, stream>>>(Wq, Wk, Wv, Wtd);

    dim3 g1(M_ / 128, NH_, 3), b1(256);
    qkv_gemm_kernel<<<g1, b1, 0, stream>>>(xbd, Wtd, bq, bk, bv, Qd, Kd, Vtd);

    dim3 g2(T_ / 128, NH_, B_), b2(256);
    attn_kernel<<<g2, b2, 0, stream>>>(Qd, Kd, Vtd, out);

    (void)in_sizes; (void)n_in; (void)out_size; (void)ws_size;
}
